// MultiheadAttention_13159779795177
// MI455X (gfx1250) — matrix operand order repaired and checked by reading, not compiled, not run on hardware
//
#include <hip/hip_runtime.h>
#include <hip/hip_bf16.h>

typedef __attribute__((ext_vector_type(16))) __bf16 v16bf;
typedef __attribute__((ext_vector_type(8)))  __bf16 v8bf;
typedef __attribute__((ext_vector_type(8)))  float  v8f;

#define WMMA_BF16(a, b, c) \
  __builtin_amdgcn_wmma_f32_16x16x32_bf16(false, (a), false, (b), (short)0, (c), false, false)

static constexpr int Bb   = 4;
static constexpr int Ss   = 2048;
static constexpr int Dd   = 1024;
static constexpr int Ee   = 1024;
static constexpr int Hh   = 16;
static constexpr int HDd  = 64;
static constexpr int Oo   = 1024;
static constexpr int Mtok = Bb * Ss;   // 8192 tokens

// ---------------------------------------------------------------------------
// CDNA5 async global->LDS copy (ASYNCcnt-tracked, no VGPR staging, no ds_store)
// ---------------------------------------------------------------------------
__device__ __forceinline__ void async_b128(unsigned lds_off, const void* gaddr) {
  asm volatile("global_load_async_to_lds_b128 %0, %1, off"
               :: "v"(lds_off), "v"(gaddr)
               : "memory");
}
__device__ __forceinline__ void wait_async0() {
  asm volatile("s_wait_asynccnt 0x0" ::: "memory");
}
__device__ __forceinline__ unsigned lds_off(const void* p) {
  return (unsigned)(size_t)p;   // low 32 bits of a generic LDS address = LDS offset
}

// ---------------------------------------------------------------------------
// DPP16 butterfly reductions within each 16-lane row (VALU-only, no LDS).
// quad_perm(1,0,3,2)=0xB1, quad_perm(2,3,0,1)=0x4E,
// row_half_mirror=0x141, row_mirror=0x140. DPP16 rows are 16 lanes, so the
// two wave halves reduce independently -- exactly the WMMA C-layout halves.
// ---------------------------------------------------------------------------
__device__ __forceinline__ float row16_max(float x) {
  int t;
  t = __builtin_amdgcn_update_dpp(0, __float_as_int(x), 0xB1, 0xf, 0xf, true);
  x = fmaxf(x, __int_as_float(t));
  t = __builtin_amdgcn_update_dpp(0, __float_as_int(x), 0x4E, 0xf, 0xf, true);
  x = fmaxf(x, __int_as_float(t));
  t = __builtin_amdgcn_update_dpp(0, __float_as_int(x), 0x141, 0xf, 0xf, true);
  x = fmaxf(x, __int_as_float(t));
  t = __builtin_amdgcn_update_dpp(0, __float_as_int(x), 0x140, 0xf, 0xf, true);
  x = fmaxf(x, __int_as_float(t));
  return x;
}

__device__ __forceinline__ float row16_sum(float x) {
  int t;
  t = __builtin_amdgcn_update_dpp(0, __float_as_int(x), 0xB1, 0xf, 0xf, true);
  x += __int_as_float(t);
  t = __builtin_amdgcn_update_dpp(0, __float_as_int(x), 0x4E, 0xf, 0xf, true);
  x += __int_as_float(t);
  t = __builtin_amdgcn_update_dpp(0, __float_as_int(x), 0x141, 0xf, 0xf, true);
  x += __int_as_float(t);
  t = __builtin_amdgcn_update_dpp(0, __float_as_int(x), 0x140, 0xf, 0xf, true);
  x += __int_as_float(t);
  return x;
}

// ---------------------------------------------------------------------------
// f32 -> bf16 conversion (bandwidth-bound pre-pass)
// ---------------------------------------------------------------------------
__global__ void cvt_f32_to_bf16(const float* __restrict__ src,
                                __bf16* __restrict__ dst, int n) {
  int i = (blockIdx.x * blockDim.x + threadIdx.x) * 4;
  if (i < n) {
    float4 f = *(const float4*)(src + i);
    dst[i + 0] = (__bf16)f.x;
    dst[i + 1] = (__bf16)f.y;
    dst[i + 2] = (__bf16)f.z;
    dst[i + 3] = (__bf16)f.w;
  }
}

// ---------------------------------------------------------------------------
// bf16 GEMM:  C[M,N] = A[M,K] * W[N,K]^T + bias,  K = 1024, block tile 128x64.
// 8 waves, each wave computes a 32x32 tile as 2x2 WMMA 16x16x32 accumulators.
// Double-buffered LDS fed by ASYNC global->LDS B128 copies (no ds_store, no
// staging VGPRs): issue into buf^1 right after the barrier, compute from buf,
// s_wait_asynccnt 0 before the next barrier publishes the writes.
// MODE 0: QKV projection -> scatter into q / k / v^T bf16 buffers.
// MODE 1: output projection -> f32 out with bias.
// ---------------------------------------------------------------------------
template <int MODE>
__global__ __launch_bounds__(256)
void gemm_bf16(const __bf16* __restrict__ A, const __bf16* __restrict__ W,
               const float* __restrict__ bias,
               __bf16* __restrict__ qout, __bf16* __restrict__ kout,
               __bf16* __restrict__ vtout, float* __restrict__ out) {
  __shared__ __align__(32) __bf16 lA[2][128 * 32];
  __shared__ __align__(32) __bf16 lB[2][64 * 32];

  const int tid  = threadIdx.x;
  const int lane = tid & 31;
  const int wave = tid >> 5;
  const int lo   = lane & 15;
  const bool hiHalf = (lane >= 16);

  const int m0 = blockIdx.y * 128;
  const int n0 = blockIdx.x * 64;
  const int wm = (wave >> 1) * 32;   // wave row offset in tile (0..96)
  const int wn = (wave & 1) * 32;    // wave col offset in tile (0 or 32)

  // Per-thread 16B-chunk assignments for the cooperative async tile fills.
  const int aRow0 = tid >> 2,          aC0 = tid & 3;           // A chunk 0
  const int aRow1 = (tid + 256) >> 2,  aC1 = tid & 3;           // A chunk 1
  const int bRow  = tid >> 2,          bC  = tid & 3;           // B chunk
  const int aP0 = ((aC0 & 1) * 2 + (aC0 >> 1)) * 8;             // K-permute
  const int aP1 = ((aC1 & 1) * 2 + (aC1 >> 1)) * 8;

  const __bf16* aSrc0 = A + (size_t)(m0 + aRow0) * 1024 + aC0 * 8;
  const __bf16* aSrc1 = A + (size_t)(m0 + aRow1) * 1024 + aC1 * 8;
  const __bf16* bSrc  = W + (size_t)(n0 + bRow) * 1024 + bC * 8;
  const unsigned aDst0 = lds_off(&lA[0][aRow0 * 32 + aP0]);
  const unsigned aDst1 = lds_off(&lA[0][aRow1 * 32 + aP1]);
  const unsigned bDst  = lds_off(&lB[0][bRow * 32 + ((bC & 1) * 2 + (bC >> 1)) * 8]);
  const unsigned aStride = lds_off(&lA[1][0]) - lds_off(&lA[0][0]);
  const unsigned bStride = lds_off(&lB[1][0]) - lds_off(&lB[0][0]);

  v8f acc[2][2] = {};

  // Prologue: async-stage k-tile 0 into buffer 0.
  async_b128(aDst0, aSrc0);
  async_b128(aDst1, aSrc1);
  async_b128(bDst, bSrc);

  for (int kt = 0; kt < 32; ++kt) {
    const int buf = kt & 1;
    wait_async0();      // our async copies into buf are done
    __syncthreads();    // ... and published to all waves
    if (kt + 1 < 32) {  // safe: barrier proved all reads of buf^1 finished
      int k0 = (kt + 1) * 32;
      unsigned o = (buf ^ 1) ? 1u : 0u;
      async_b128(aDst0 + o * aStride, aSrc0 + k0);
      async_b128(aDst1 + o * aStride, aSrc1 + k0);
      async_b128(bDst + o * bStride, bSrc + k0);
    }

    v16bf af[2], bfrag[2];
#pragma unroll
    for (int mi = 0; mi < 2; ++mi)
      af[mi] = *(const v16bf*)&lA[buf][(wm + mi * 16 + lo) * 32 + (hiHalf ? 16 : 0)];
#pragma unroll
    for (int ni = 0; ni < 2; ++ni)
      bfrag[ni] = *(const v16bf*)&lB[buf][(wn + ni * 16 + lo) * 32 + (hiHalf ? 16 : 0)];
#pragma unroll
    for (int mi = 0; mi < 2; ++mi)
#pragma unroll
      for (int ni = 0; ni < 2; ++ni)
        acc[mi][ni] = WMMA_BF16(af[mi], bfrag[ni], acc[mi][ni]);
  }

  // Epilogue. C layout: VGPR i -> row i (lanes 0-15) / row i+8 (lanes 16-31),
  // col = lane&15.
#pragma unroll
  for (int mi = 0; mi < 2; ++mi)
#pragma unroll
    for (int ni = 0; ni < 2; ++ni)
#pragma unroll
      for (int i = 0; i < 8; ++i) {
        int row = m0 + wm + mi * 16 + i + (hiHalf ? 8 : 0);
        int col = n0 + wn + ni * 16 + lo;
        float v = acc[mi][ni][i] + bias[col];
        if (MODE == 0) {
          // col in [0,3072): per token layout is H groups of [q(64)|k(64)|v(64)]
          int h    = col / 192;
          int r    = col % 192;
          int part = r >> 6;
          int hd   = r & 63;
          int b    = row >> 11;        // row / S
          int s    = row & 2047;       // row % S
          size_t bh = (size_t)(b * Hh + h);
          if (part == 0)
            qout[(bh * Ss + s) * HDd + hd] = (__bf16)v;
          else if (part == 1)
            kout[(bh * Ss + s) * HDd + hd] = (__bf16)v;
          else
            vtout[(bh * HDd + hd) * Ss + s] = (__bf16)v;   // V transposed
        } else {
          out[(size_t)row * Oo + col] = v;
        }
      }
}

// ---------------------------------------------------------------------------
// Flash attention. Block = 8 waves = 128 query rows of one (b,h).
// Each wave: 16 q rows, online softmax over key tiles of 32, HD=64 output.
// K/V tiles double-buffered through LDS via async B128 copies; softmax via
// DPP16; 1/sqrt(HD) folded into the Q fragments (exact power of two in bf16).
// ---------------------------------------------------------------------------
__global__ __launch_bounds__(256)
void attn_kernel(const __bf16* __restrict__ q, const __bf16* __restrict__ k,
                 const __bf16* __restrict__ vt, __bf16* __restrict__ vals) {
  __shared__ __align__(32) __bf16 lK[2][32 * 64];   // key tile [32 keys][64 hd]
  __shared__ __align__(32) __bf16 lV[2][64 * 32];   // V^T tile [64 hd][32 keys]
  __shared__ __align__(32) __bf16 lP[8][16 * 32];   // per-wave P scratch

  const int bh   = blockIdx.x;           // b*H + h
  const int tid  = threadIdx.x;
  const int lane = tid & 31;
  const int wave = tid >> 5;
  const int lo   = lane & 15;
  const bool hiHalf = (lane >= 16);

  const int qrow0 = blockIdx.y * 128 + wave * 16;

  // Cooperative K/V async staging assignments (one 16B chunk each per tile).
  const int kRow = tid >> 3, kC = tid & 7;   // K: 32 rows x 8 chunks
  const int vRow = tid >> 2, vC = tid & 3;   // V^T: 64 rows x 4 chunks
  const __bf16* kSrc = k + ((size_t)bh * Ss + kRow) * HDd + kC * 8;
  const __bf16* vSrc = vt + ((size_t)bh * HDd + vRow) * Ss + vC * 8;
  const unsigned kDst = lds_off(&lK[0][kRow * 64 + kC * 8]);
  const unsigned vDst = lds_off(&lV[0][vRow * 32 + vC * 8]);
  const unsigned kStride = lds_off(&lK[1][0]) - lds_off(&lK[0][0]);
  const unsigned vStride = lds_off(&lV[1][0]) - lds_off(&lV[0][0]);

  // Q fragments (16 rows x 64 hd -> two 16x32 A fragments), direct from
  // global, pre-scaled by 1/sqrt(64) = 0.125 (exact in bf16).
  v16bf qf[2];
#pragma unroll
  for (int f = 0; f < 2; ++f) {
    const __bf16* p =
        q + ((size_t)bh * Ss + qrow0 + lo) * HDd + f * 32 + (hiHalf ? 16 : 0);
    v8bf a0 = *(const v8bf*)p;          // k {0..7} (lo) / {16..23} (hi)
    v8bf a1 = *(const v8bf*)(p + 8);    // k {8..15} / {24..31}
#pragma unroll
    for (int i = 0; i < 8; ++i) {
      qf[f][i]     = (__bf16)((float)a0[i] * 0.125f);
      qf[f][8 + i] = (__bf16)((float)a1[i] * 0.125f);
    }
  }

  v8f oacc[4] = {};
  float mrun[8], lrun[8];
#pragma unroll
  for (int i = 0; i < 8; ++i) { mrun[i] = -1e30f; lrun[i] = 0.f; }

  constexpr int NT = Ss / 32;  // 64 key tiles

  // Prologue: async-stage key tile 0 into buffer 0.
  async_b128(kDst, kSrc);
  async_b128(vDst, vSrc);

  for (int kt = 0; kt < NT; ++kt) {
    const int buf = kt & 1;
    wait_async0();      // our async copies into buf are done
    __syncthreads();    // ... and published to all waves
    if (kt + 1 < NT) {  // stage next tile into the other buffer
      unsigned o = (buf ^ 1) ? 1u : 0u;
      async_b128(kDst + o * kStride, kSrc + (size_t)(kt + 1) * 32 * HDd);
      async_b128(vDst + o * vStride, vSrc + (kt + 1) * 32);
    }

    // S = Q*K^T for 32 keys as two 16x16 tiles, K-depth 64 in two WMMA steps.
    v8f s0 = {}, s1 = {};
#pragma unroll
    for (int f = 0; f < 2; ++f) {
      v16bf b0 = *(const v16bf*)&lK[buf][lo * 64 + f * 32 + (hiHalf ? 16 : 0)];
      s0 = WMMA_BF16(qf[f], b0, s0);
      v16bf b1 = *(const v16bf*)&lK[buf][(16 + lo) * 64 + f * 32 + (hiHalf ? 16 : 0)];
      s1 = WMMA_BF16(qf[f], b1, s1);
    }

    // Online softmax. Row r lives in VGPR i (=r or r-8) across a 16-lane half;
    // DPP16 butterflies reduce each half independently.
    float fac[8];
#pragma unroll
    for (int i = 0; i < 8; ++i) {
      float t = row16_max(fmaxf(s0[i], s1[i]));
      float nm = fmaxf(mrun[i], t);
      fac[i] = __expf(mrun[i] - nm);
      mrun[i] = nm;
      s0[i] = __expf(s0[i] - nm);
      s1[i] = __expf(s1[i] - nm);
      float ps = row16_sum(s0[i] + s1[i]);
      lrun[i] = lrun[i] * fac[i] + ps;
    }
    // Rescale output accumulators.
#pragma unroll
    for (int c = 0; c < 4; ++c)
#pragma unroll
      for (int i = 0; i < 8; ++i) oacc[c][i] = oacc[c][i] * fac[i];

    // P (C-layout f32) -> bf16 in wave-private LDS, stored in A-fragment
    // contiguous-K order so readback is one contiguous 32B load per lane.
    __bf16* pw = lP[wave];
    const int posa = lo;        // key j = lo
    const int posb = 16 + lo;   // key j = 16+lo
#pragma unroll
    for (int i = 0; i < 8; ++i) {
      int row = i + (hiHalf ? 8 : 0);
      pw[row * 32 + posa] = (__bf16)s0[i];
      pw[row * 32 + posb] = (__bf16)s1[i];
    }
    v16bf pa = *(const v16bf*)&pw[lo * 32 + (hiHalf ? 16 : 0)];

    // O += P * V  (4 hd-chunks of 16)
#pragma unroll
    for (int c = 0; c < 4; ++c) {
      v16bf bv = *(const v16bf*)&lV[buf][(c * 16 + lo) * 32 + (hiHalf ? 16 : 0)];
      oacc[c] = WMMA_BF16(pa, bv, oacc[c]);
    }
  }

  // Epilogue: normalize and write vals[b, s, h*64 + hd] as bf16.
  const int b = bh >> 4;
  const int h = bh & 15;
#pragma unroll
  for (int c = 0; c < 4; ++c)
#pragma unroll
    for (int i = 0; i < 8; ++i) {
      int srow = qrow0 + i + (hiHalf ? 8 : 0);
      float v = oacc[c][i] / lrun[i];
      vals[((size_t)(b * Ss + srow)) * Ee + h * 64 + c * 16 + lo] = (__bf16)v;
    }
}

// ---------------------------------------------------------------------------
// Host launcher
// ---------------------------------------------------------------------------
extern "C" void kernel_launch(void* const* d_in, const int* in_sizes, int n_in,
                              void* d_out, int out_size, void* d_ws, size_t ws_size,
                              hipStream_t stream) {
  const float* x     = (const float*)d_in[0];
  const float* qkv_w = (const float*)d_in[1];
  const float* qkv_b = (const float*)d_in[2];
  const float* o_w   = (const float*)d_in[3];
  const float* o_b   = (const float*)d_in[4];
  float* out = (float*)d_out;

  // Workspace carve (all bf16): ~88 MB total.
  char* p = (char*)d_ws;
  __bf16* xb    = (__bf16*)p; p += (size_t)Mtok * Dd * 2;     // 16 MB
  __bf16* wqkv  = (__bf16*)p; p += (size_t)3 * Ee * Dd * 2;   // 6 MB
  __bf16* wo    = (__bf16*)p; p += (size_t)Oo * Ee * 2;       // 2 MB
  __bf16* qb    = (__bf16*)p; p += (size_t)Mtok * Ee * 2;     // 16 MB
  __bf16* kb    = (__bf16*)p; p += (size_t)Mtok * Ee * 2;     // 16 MB
  __bf16* vtb   = (__bf16*)p; p += (size_t)Mtok * Ee * 2;     // 16 MB
  __bf16* valsb = (__bf16*)p; p += (size_t)Mtok * Ee * 2;     // 16 MB

  // 1) bf16 conversion pre-passes
  {
    int n = Mtok * Dd;
    cvt_f32_to_bf16<<<dim3((n / 4 + 255) / 256), 256, 0, stream>>>(x, xb, n);
  }
  {
    int n = 3 * Ee * Dd;
    cvt_f32_to_bf16<<<dim3((n / 4 + 255) / 256), 256, 0, stream>>>(qkv_w, wqkv, n);
  }
  {
    int n = Oo * Ee;
    cvt_f32_to_bf16<<<dim3((n / 4 + 255) / 256), 256, 0, stream>>>(o_w, wo, n);
  }

  // 2) QKV projection GEMM: [8192,1024] x [3072,1024]^T -> q/k/v^T (bf16)
  gemm_bf16<0><<<dim3(3 * Ee / 64, Mtok / 128), 256, 0, stream>>>(
      xb, wqkv, qkv_b, qb, kb, vtb, nullptr);

  // 3) Flash attention -> vals (bf16)
  attn_kernel<<<dim3(Bb * Hh, Ss / 128), 256, 0, stream>>>(qb, kb, vtb, valsb);

  // 4) Output projection GEMM: [8192,1024] x [1024,1024]^T + o_b -> f32 out
  gemm_bf16<1><<<dim3(Oo / 64, Mtok / 128), 256, 0, stream>>>(
      valsb, wo, o_b, nullptr, nullptr, nullptr, out);
}
